// MultiEmbeder_83717502534009
// MI455X (gfx1250) — matrix operand order repaired and checked by reading, not compiled, not run on hardware
//
#include <hip/hip_runtime.h>
#include <hip/hip_bf16.h>

// ---------------------------------------------------------------------------
// Types for CDNA5 WMMA bf16 path + TDM descriptors
// ---------------------------------------------------------------------------
typedef __bf16 bf16_t;
typedef __attribute__((ext_vector_type(16))) __bf16 bfx16;
typedef __attribute__((ext_vector_type(8)))  float  fx8;
typedef __attribute__((ext_vector_type(4)))  unsigned int u32x4;
typedef __attribute__((ext_vector_type(8)))  int i32x8;
typedef __attribute__((ext_vector_type(4)))  int i32x4;

union FragU { uint4 u[2]; bfx16 v; };

// Load 16 bf16 (two 16-byte chunks) into a WMMA fragment register group.
__device__ __forceinline__ bfx16 ld_frag_b16(const bf16_t* p0, const bf16_t* p1) {
    FragU f;
    f.u[0] = *(const uint4*)p0;
    f.u[1] = *(const uint4*)p1;
    return f.v;
}

// ---------------------------------------------------------------------------
// TDM: DMA a [512 rows x 32 K] bf16 tile of Wt (row stride Kdim elements,
// starting at column kc) into LDS at byte offset lds_off, with LDS padding
// of 4 DWORDs per 16 DWORDs (row stride in LDS = 64B data + 16B pad = 80B).
// D# per CDNA5 ISA ch.8: group0 = {count/lds_addr/global_addr/type},
// group1 = {data_size, pad cfg, tensor dims, tile dims, strides}.
// ---------------------------------------------------------------------------
__device__ __forceinline__ void tdm_load_wt_chunk(const bf16_t* Wt, int kc,
                                                  unsigned lds_off, int Kdim) {
    unsigned long long ga =
        (unsigned long long)(uintptr_t)Wt + (unsigned long long)kc * 2ull;
    u32x4 g0;
    g0[0] = 1u;                                   // count=1 (valid), user mode
    g0[1] = lds_off;                              // lds_addr [63:32]
    g0[2] = (unsigned)(ga & 0xffffffffull);       // global_addr [95:64]
    g0[3] = (unsigned)((ga >> 32) & 0x01ffffffull)// global_addr [120:96]
          | (2u << 30);                           // type=2 ("image") [127:126]
    i32x8 g1;
    g1[0] = (1 << 16)        // data_size = 1 -> 2 bytes (bf16)
          | (1 << 20)        // pad_enable
          | (3 << 22)        // pad_interval = 16 DWORDs (64B)
          | (3 << 25);       // pad_amount  = 4 DWORDs (16B)
    g1[1] = (Kdim & 0xffff) << 16;                  // tensor_dim0[15:0] @ bits 63:48
    g1[2] = (512 << 16) | ((Kdim >> 16) & 0xffff);  // tensor_dim1=512, tensor_dim0 hi
    g1[3] = (32 << 16);                             // tile_dim0 = 32 @ bits 127:112
    g1[4] = 512;                                    // tile_dim1 = 512 rows
    g1[5] = Kdim;                                   // tensor_dim0_stride[31:0]
    g1[6] = 0;
    g1[7] = 0;
    i32x4 z = {0, 0, 0, 0};
#if defined(__clang_major__) && (__clang_major__ >= 23)
    i32x8 z8 = {0, 0, 0, 0, 0, 0, 0, 0};
    __builtin_amdgcn_tensor_load_to_lds(g0, g1, z, z, z8, 0);
#else
    __builtin_amdgcn_tensor_load_to_lds(g0, g1, z, z, 0);
#endif
}

// ---------------------------------------------------------------------------
// Kernel 0: W [K,N] f32 row-major  ->  Wt [N,K] bf16 (B-fragment friendly)
// ---------------------------------------------------------------------------
__global__ void wconv_kernel(const float* __restrict__ W, bf16_t* __restrict__ Wt,
                             int K, int N) {
    int idx = blockIdx.x * 256 + threadIdx.x;
    if (idx < K * N) {
        int k = idx / N;
        int n = idx - k * N;
        Wt[(size_t)n * K + k] = (bf16_t)W[idx];
    }
}

// ---------------------------------------------------------------------------
// Kernel 1: logits[m] = tanh(X[m,:] @ W + b) . v + c     (K = H = 512 fixed)
// Block = 256 threads (8 wave32), 16 rows per block, full 512 cols.
// Wave w handles col group [w*64, w*64+64) = 4 N-tiles; W K-chunks are
// TDM-DMA'd into LDS (with padding) and shared by all waves.
// ---------------------------------------------------------------------------
__launch_bounds__(256)
__global__ void logits_wmma_kernel(const float* __restrict__ X,
                                   const bf16_t* __restrict__ Wt,   // [512][512] bf16
                                   const float* __restrict__ bias,
                                   const float* __restrict__ vvec,
                                   const float* __restrict__ cptr,
                                   float* __restrict__ logits) {
    constexpr int K    = 512;
    constexpr int WROW = 40;   // 32 data + 8 pad bf16 = 80B row (TDM pad cfg)
    __shared__ __align__(16) bf16_t As[16][K + 8];   // 16.6 KB
    __shared__ __align__(16) bf16_t Wbuf[512][WROW]; // 40 KB
    __shared__ float logit_s[16];

    const int    tid     = threadIdx.x;
    const size_t rowbase = (size_t)blockIdx.x * 16;

    if (tid < 16) logit_s[tid] = 0.f;

    // Stage 16 rows of X (f32 -> bf16) into LDS.
    for (int i = tid; i < 16 * (K / 4); i += 256) {
        int r  = i >> 7;            // / (K/4)
        int c4 = (i & 127) * 4;
        const float4 f = *(const float4*)(X + (rowbase + r) * K + c4);
        As[r][c4 + 0] = (bf16_t)f.x;
        As[r][c4 + 1] = (bf16_t)f.y;
        As[r][c4 + 2] = (bf16_t)f.z;
        As[r][c4 + 3] = (bf16_t)f.w;
    }

    const int wave     = tid >> 5;
    const int lane     = tid & 31;
    const int half     = lane >> 4;
    const int lr       = lane & 15;
    const int ncolbase = wave * 64;
    const unsigned wlds = (unsigned)(uintptr_t)(&Wbuf[0][0]);

    fx8 acc[4] = {};
    const bf16_t* arow = &As[lr][0];

    for (int kc = 0; kc < K; kc += 32) {
        __syncthreads();                 // previous Wbuf chunk consumed / As staged
        if (wave == 0) {
            tdm_load_wt_chunk(Wt, kc, wlds, K);
            __builtin_amdgcn_s_wait_tensorcnt(0);
        }
        __syncthreads();                 // Wbuf chunk ready for all waves

        // A 16x32 bf16 fragment per ISA layout:
        //  lanes 0-15: K=[kc..kc+7],[kc+16..kc+23]; lanes 16-31: +8
        bfx16 afrag = ld_frag_b16(arow + kc + half * 8,
                                  arow + kc + 16 + half * 8);
#pragma unroll
        for (int t = 0; t < 4; ++t) {
            // B 32x16 fragment from padded LDS chunk: lane row n, K same split as A.
            const bf16_t* brow = &Wbuf[ncolbase + t * 16 + lr][half * 8];
            bfx16 bfrag = ld_frag_b16(brow, brow + 16);
            acc[t] = __builtin_amdgcn_wmma_f32_16x16x32_bf16(
                false, afrag, false, bfrag, (short)0, acc[t], false, false);
        }
    }

    // Epilogue: tanh(acc + b[n]) * v[n], reduce over this wave's 64 cols.
    float part[8] = {0.f, 0.f, 0.f, 0.f, 0.f, 0.f, 0.f, 0.f};
#pragma unroll
    for (int t = 0; t < 4; ++t) {
        const int   n  = ncolbase + t * 16 + lr;
        const float bn = bias[n];
        const float vn = vvec[n];
#pragma unroll
        for (int r = 0; r < 8; ++r)
            part[r] += tanhf(acc[t][r] + bn) * vn;
    }
    // Reduce across the 16 lanes sharing the same output row (xor bits 0..3).
#pragma unroll
    for (int r = 0; r < 8; ++r) {
#pragma unroll
        for (int m = 1; m < 16; m <<= 1)
            part[r] += __shfl_xor(part[r], m, 32);
    }
    // C/D layout: VGPR r holds M = r + 8*half (N = lr).
    if (lr == 0) {
        const int mloc = half * 8;
#pragma unroll
        for (int r = 0; r < 8; ++r)
            atomicAdd(&logit_s[mloc + r], part[r]);   // ds_add_f32 across waves
    }
    __syncthreads();
    if (tid < 16)
        logits[rowbase + tid] = logit_s[tid] + cptr[0];
}

// ---------------------------------------------------------------------------
// Kernel 2: per-segment masked softmax attention pooling.
// ---------------------------------------------------------------------------
template <bool MASKED>
__launch_bounds__(256)
__global__ void softmax_pool_kernel(const float* __restrict__ logits,
                                    const float* __restrict__ X,
                                    const long long* __restrict__ len,
                                    float* __restrict__ outc,
                                    int S, int H, int cstride, int coff) {
    __shared__ float w[512];
    __shared__ float red[256];
    const int b   = blockIdx.x;
    const int tid = threadIdx.x;
    const long long L = MASKED ? len[b] : (long long)S;

    float lmax = -3.0e38f;
    for (int s = tid; s < S; s += 256) {
        float l = logits[(size_t)b * S + s];
        if (MASKED && (long long)s >= L) l = -1.0e30f;
        w[s] = l;
        lmax = fmaxf(lmax, l);
    }
    red[tid] = lmax; __syncthreads();
    for (int st = 128; st > 0; st >>= 1) {
        if (tid < st) red[tid] = fmaxf(red[tid], red[tid + st]);
        __syncthreads();
    }
    const float mx = red[0];
    __syncthreads();

    float lsum = 0.f;
    for (int s = tid; s < S; s += 256) {
        float e = __expf(w[s] - mx);
        w[s] = e;
        lsum += e;
    }
    red[tid] = lsum; __syncthreads();
    for (int st = 128; st > 0; st >>= 1) {
        if (tid < st) red[tid] += red[tid + st];
        __syncthreads();
    }
    const float inv = 1.0f / red[0];
    __syncthreads();

    for (int h = tid; h < H; h += 256) {
        float acc = 0.f;
        const float* Xb = X + ((size_t)b * S) * H + h;
        for (int s = 0; s < S; ++s) acc += w[s] * Xb[(size_t)s * H];
        outc[(size_t)b * cstride + coff + h] = acc * inv;
    }
}

// ---------------------------------------------------------------------------
// Kernel 3: sigmoid-gated masked pooling (CFG branch).
// ---------------------------------------------------------------------------
__launch_bounds__(256)
__global__ void gate_pool_kernel(const float* __restrict__ logits,
                                 const float* __restrict__ X,
                                 const int* __restrict__ mask,
                                 float* __restrict__ outc,
                                 int N, int H, int cstride, int coff) {
    __shared__ float g[512];
    const int b   = blockIdx.x;
    const int tid = threadIdx.x;
    for (int n = tid; n < N; n += 256) {
        float l = logits[(size_t)b * N + n];
        g[n] = mask[(size_t)b * N + n] ? (1.0f / (1.0f + __expf(-l))) : 0.f;
    }
    __syncthreads();
    for (int h = tid; h < H; h += 256) {
        float acc = 0.f;
        const float* Xb = X + ((size_t)b * N) * H + h;
        for (int n = 0; n < N; ++n) acc += g[n] * Xb[(size_t)n * H];
        outc[(size_t)b * cstride + coff + h] = acc;
    }
}

// ---------------------------------------------------------------------------
// Kernel 4: fusion GEMM  out[128,512] = tanh(concat[128,1536] @ W_fuse + b).
// A converted f32->bf16 on the fly (A is tiny, L2-resident), K = 1536.
// ---------------------------------------------------------------------------
__launch_bounds__(256)
__global__ void fuse_wmma_kernel(const float* __restrict__ A,      // [rows][1536]
                                 const bf16_t* __restrict__ Wt,    // [512][1536] bf16
                                 const float* __restrict__ bias,
                                 float* __restrict__ out) {        // [rows][512]
    constexpr int K = 1536;
    const int    tid      = threadIdx.x;
    const int    wave     = tid >> 5;
    const int    lane     = tid & 31;
    const int    half     = lane >> 4;
    const int    lr       = lane & 15;
    const int    mtile    = wave & 1;
    const int    ncolbase = (wave >> 1) * 128;
    const size_t rowbase  = (size_t)blockIdx.x * 32;

    fx8 acc[8] = {};
    const float* arow = A + (rowbase + mtile * 16 + lr) * K;

    for (int kc = 0; kc < K; kc += 32) {
        FragU fa;
        const float* p0 = arow + kc + half * 8;
        const float* p1 = arow + kc + 16 + half * 8;
#pragma unroll
        for (int i = 0; i < 8; ++i) fa.v[i] = (bf16_t)p0[i];
#pragma unroll
        for (int i = 0; i < 8; ++i) fa.v[8 + i] = (bf16_t)p1[i];
        bfx16 afrag = fa.v;
#pragma unroll
        for (int t = 0; t < 8; ++t) {
            const bf16_t* brow = Wt + (size_t)(ncolbase + t * 16 + lr) * K
                                    + kc + half * 8;
            bfx16 bfrag = ld_frag_b16(brow, brow + 16);
            acc[t] = __builtin_amdgcn_wmma_f32_16x16x32_bf16(
                false, afrag, false, bfrag, (short)0, acc[t], false, false);
        }
    }
#pragma unroll
    for (int t = 0; t < 8; ++t) {
        const int   n  = ncolbase + t * 16 + lr;
        const float bn = bias[n];
#pragma unroll
        for (int r = 0; r < 8; ++r) {
            const size_t m = rowbase + mtile * 16 + half * 8 + r;
            out[m * 512 + n] = tanhf(acc[t][r] + bn);
        }
    }
}

// ---------------------------------------------------------------------------
// Host-side launcher
// ---------------------------------------------------------------------------
extern "C" void kernel_launch(void* const* d_in, const int* in_sizes, int n_in,
                              void* d_out, int out_size, void* d_ws, size_t ws_size,
                              hipStream_t stream) {
    (void)n_in; (void)out_size; (void)ws_size;

    const float*     tok_feat = (const float*)d_in[0];
    const float*     ast_h    = (const float*)d_in[1];
    const float*     cfg_feat = (const float*)d_in[2];
    const float*     W_tok    = (const float*)d_in[3];
    const float*     b_tok    = (const float*)d_in[4];
    const float*     v_tok    = (const float*)d_in[5];
    const float*     c_tok    = (const float*)d_in[6];
    const float*     W_ast    = (const float*)d_in[7];
    const float*     b_ast    = (const float*)d_in[8];
    const float*     v_ast    = (const float*)d_in[9];
    const float*     c_ast    = (const float*)d_in[10];
    const float*     W_cfg    = (const float*)d_in[11];
    const float*     b_cfg    = (const float*)d_in[12];
    const float*     v_cfg    = (const float*)d_in[13];
    const float*     c_cfg    = (const float*)d_in[14];
    const float*     W_fuse   = (const float*)d_in[15];
    const float*     b_fuse   = (const float*)d_in[16];
    const long long* tok_len  = (const long long*)d_in[17];
    /* d_in[18] = node_seg: contiguous equal segments (T/B each), used implicitly */
    const int*       cfg_mask = (const int*)d_in[19];

    const int B = in_sizes[17];           // 128
    const int H = in_sizes[4];            // 512 (kernels assume 512)
    const int S = in_sizes[0] / (B * H);  // 512
    const int N = in_sizes[2] / (B * H);  // 256
    const int T = in_sizes[1] / H;        // 40960
    const int G = T / B;                  // 320 nodes per segment

    // Workspace carve-out (256B aligned slabs).
    size_t off = 0;
    char*  wsb = (char*)d_ws;
    auto take = [&](size_t bytes) -> void* {
        void* p = wsb + off;
        off += (bytes + 255) & ~(size_t)255;
        return p;
    };
    bf16_t* Wt_tok  = (bf16_t*)take((size_t)H * H * 2);
    bf16_t* Wt_ast  = (bf16_t*)take((size_t)H * H * 2);
    bf16_t* Wt_cfg  = (bf16_t*)take((size_t)H * H * 2);
    bf16_t* Wt_fuse = (bf16_t*)take((size_t)3 * H * H * 2);
    float*  lg_tok  = (float*)take((size_t)B * S * 4);
    float*  lg_ast  = (float*)take((size_t)T * 4);
    float*  lg_cfg  = (float*)take((size_t)B * N * 4);
    float*  concat  = (float*)take((size_t)B * 3 * H * 4);

    float* out = (float*)d_out;
    const int CS = 3 * H;   // concat row stride

    // 0) weight convert/transpose (f32 -> bf16 W^T)
    wconv_kernel<<<(H * H + 255) / 256, 256, 0, stream>>>(W_tok, Wt_tok, H, H);
    wconv_kernel<<<(H * H + 255) / 256, 256, 0, stream>>>(W_ast, Wt_ast, H, H);
    wconv_kernel<<<(H * H + 255) / 256, 256, 0, stream>>>(W_cfg, Wt_cfg, H, H);
    wconv_kernel<<<(3 * H * H + 255) / 256, 256, 0, stream>>>(W_fuse, Wt_fuse, 3 * H, H);

    // 1) logits (TDM-staged W + WMMA GEMM + fused tanh/dot epilogue)
    logits_wmma_kernel<<<(B * S) / 16, 256, 0, stream>>>(tok_feat, Wt_tok, b_tok, v_tok, c_tok, lg_tok);
    logits_wmma_kernel<<<T / 16,       256, 0, stream>>>(ast_h,    Wt_ast, b_ast, v_ast, c_ast, lg_ast);
    logits_wmma_kernel<<<(B * N) / 16, 256, 0, stream>>>(cfg_feat, Wt_cfg, b_cfg, v_cfg, c_cfg, lg_cfg);

    // 2) pooling
    softmax_pool_kernel<true ><<<B, 256, 0, stream>>>(lg_tok, tok_feat, tok_len, concat, S, H, CS, 0);
    softmax_pool_kernel<false><<<B, 256, 0, stream>>>(lg_ast, ast_h,    nullptr, concat, G, H, CS, H);
    gate_pool_kernel<<<B, 256, 0, stream>>>(lg_cfg, cfg_feat, cfg_mask, concat, N, H, CS, 2 * H);

    // 3) fusion GEMM -> d_out [B, H]
    fuse_wmma_kernel<<<B / 32, 256, 0, stream>>>(concat, Wt_fuse, b_fuse, out);
}
